// GraphEncoder_52810917872238
// MI455X (gfx1250) — matrix operand order repaired and checked by reading, not compiled, not run on hardware
//
#include <hip/hip_runtime.h>
#include <hip/hip_bf16.h>

typedef __attribute__((ext_vector_type(16))) __bf16 v16bf;
typedef __attribute__((ext_vector_type(8)))  float  v8f;

union FragA { uint4 u[2]; v16bf v; };

#define NNODE 400
#define CDIM  256
#define HD    64
#define NPAD  416

__device__ __forceinline__ float wred_sum(float v) {
#pragma unroll
  for (int o = 16; o > 0; o >>= 1) v += __shfl_xor(v, o, 32);
  return v;
}
__device__ __forceinline__ float wred_max(float v) {
#pragma unroll
  for (int o = 16; o > 0; o >>= 1) v = fmaxf(v, __shfl_xor(v, o, 32));
  return v;
}
__device__ __forceinline__ unsigned pk2(float lo, float hi) {
  union { __bf16 h[2]; unsigned u; } p;
  p.h[0] = (__bf16)lo;
  p.h[1] = (__bf16)hi;
  return p.u;       // v_cvt_pk_bf16_f32
}
__device__ __forceinline__ v8f wmma_bf16(v16bf a, v16bf b, v8f c) {
  return __builtin_amdgcn_wmma_f32_16x16x32_bf16(false, a, false, b, (short)0, c,
                                                 false, false);
}
// stage 32 contiguous floats (8x float4) -> 32 packed bf16 in LDS (4x b128)
__device__ __forceinline__ void stage32(const float* __restrict__ g, __bf16* s) {
  const float4* g4 = (const float4*)g;
  float4 f[8];
#pragma unroll
  for (int i = 0; i < 8; ++i) f[i] = g4[i];
  uint4* s4 = (uint4*)s;
#pragma unroll
  for (int i = 0; i < 4; ++i) {
    uint4 u;
    u.x = pk2(f[2 * i].x, f[2 * i].y);
    u.y = pk2(f[2 * i].z, f[2 * i].w);
    u.z = pk2(f[2 * i + 1].x, f[2 * i + 1].y);
    u.w = pk2(f[2 * i + 1].z, f[2 * i + 1].w);
    s4[i] = u;
  }
}

// ---------------------------------------------------------------------------
// Batched GEMM: C[b] = A[b] @ B[b] (+ bias[n]) (+ res[b])
// A: [M,K] ld=K, B: [K,N] ld=N, C/res: [M,N] ld=N, all row-major f32.
// 128 threads = 4 waves; 64x64 block tile; each wave owns a 32x32 subtile
// (2x2 WMMA accumulators). K stepped by 32 through LDS in bf16.
// Edge tiles use a three-way split (vector / zero / scalar) so the K-tail of
// K=400 GEMMs and the N=400 tail tile stay vectorized.
// ---------------------------------------------------------------------------
__global__ __launch_bounds__(128) void gemm_wmma(
    const float* __restrict__ A, const float* __restrict__ B,
    const float* __restrict__ bias, const float* __restrict__ res,
    float* __restrict__ C, int M, int N, int K,
    long long sA, long long sB, long long sC)
{
  const int bz = blockIdx.z;
  A += (long long)bz * sA;
  B += (long long)bz * sB;
  C += (long long)bz * sC;
  const float* R = res ? (res + (long long)bz * sC) : nullptr;

  const int tm = blockIdx.y * 64, tn = blockIdx.x * 64;

  __shared__ __align__(16) __bf16 As[64 * 40];   // As[row][k], 80B row pitch
  __shared__ __align__(16) __bf16 Bs[64 * 40];   // Bs[col][k] (transposed)

  const int t = threadIdx.x;
  const int wave = t >> 5, lane = t & 31;
  const int wm = (wave >> 1) * 32, wn = (wave & 1) * 32;
  const int row16 = lane & 15, kh = lane >> 4;

  const int ar  = t >> 1;          // A row 0..63
  const int aks = (t & 1) << 4;    // A k-segment 0/16
  const int bnc = t & 63;          // B col 0..63
  const int bks = (t >> 6) << 4;   // B k-segment 0/16

  const v8f vz = {0.f, 0.f, 0.f, 0.f, 0.f, 0.f, 0.f, 0.f};
  v8f acc[2][2];
  acc[0][0] = vz; acc[0][1] = vz; acc[1][0] = vz; acc[1][1] = vz;

  const int nk = (K + 31) >> 5;
  const bool mfull = (tm + 64 <= M);
  const bool nfull = (tn + 64 <= N);

  for (int ks = 0; ks < nk; ++ks) {
    const int k0 = ks << 5;
    const bool kfull = (k0 + 32 <= K);

    // ---------------- A tile ----------------
    if (mfull && kfull) {
      const float* ga = A + (long long)(tm + ar) * K + (k0 + aks);
      if (k0 + 64 <= K) __builtin_prefetch(ga + 32, 0, 0);
      stage32(ga, &As[ar * 40 + aks]);
    } else {
      const int gm = tm + ar;
      const int kbeg = k0 + aks;
      uint4* s4 = (uint4*)&As[ar * 40 + aks];
      if (gm < M && kbeg + 16 <= K) {          // segment fully valid
        const float4* g4 = (const float4*)(A + (long long)gm * K + kbeg);
        float4 f0 = g4[0], f1 = g4[1], f2 = g4[2], f3 = g4[3];
        uint4 u0, u1;
        u0.x = pk2(f0.x, f0.y); u0.y = pk2(f0.z, f0.w);
        u0.z = pk2(f1.x, f1.y); u0.w = pk2(f1.z, f1.w);
        u1.x = pk2(f2.x, f2.y); u1.y = pk2(f2.z, f2.w);
        u1.z = pk2(f3.x, f3.y); u1.w = pk2(f3.z, f3.w);
        s4[0] = u0; s4[1] = u1;
      } else if (gm >= M || kbeg >= K) {       // segment fully out -> zeros
        uint4 z = {0u, 0u, 0u, 0u};
        s4[0] = z; s4[1] = z;
      } else {                                 // ragged (unused by our shapes)
#pragma unroll
        for (int j = 0; j < 16; ++j) {
          int gk = kbeg + j;
          float v = (gk < K) ? A[(long long)gm * K + gk] : 0.f;
          As[ar * 40 + aks + j] = (__bf16)v;
        }
      }
    }

    // ---------------- B tile (column gather, stored transposed) -------------
    if (nfull && kfull) {
      const float* gb = B + (long long)(k0 + bks) * N + (tn + bnc);
      if (k0 + 64 <= K) __builtin_prefetch(gb + 32LL * N, 0, 0);
      float f[16];
#pragma unroll
      for (int j = 0; j < 16; ++j) f[j] = gb[(long long)j * N];
      uint4 u0, u1;
      u0.x = pk2(f[0], f[1]);   u0.y = pk2(f[2], f[3]);
      u0.z = pk2(f[4], f[5]);   u0.w = pk2(f[6], f[7]);
      u1.x = pk2(f[8], f[9]);   u1.y = pk2(f[10], f[11]);
      u1.z = pk2(f[12], f[13]); u1.w = pk2(f[14], f[15]);
      uint4* s4 = (uint4*)&Bs[bnc * 40 + bks];
      s4[0] = u0; s4[1] = u1;
    } else {
      const int gn = tn + bnc;
      const int kbeg = k0 + bks;
      uint4* s4 = (uint4*)&Bs[bnc * 40 + bks];
      if (gn < N && kbeg + 16 <= K) {
        const float* gb = B + (long long)kbeg * N + gn;
        float f[16];
#pragma unroll
        for (int j = 0; j < 16; ++j) f[j] = gb[(long long)j * N];
        uint4 u0, u1;
        u0.x = pk2(f[0], f[1]);   u0.y = pk2(f[2], f[3]);
        u0.z = pk2(f[4], f[5]);   u0.w = pk2(f[6], f[7]);
        u1.x = pk2(f[8], f[9]);   u1.y = pk2(f[10], f[11]);
        u1.z = pk2(f[12], f[13]); u1.w = pk2(f[14], f[15]);
        s4[0] = u0; s4[1] = u1;
      } else if (gn >= N || kbeg >= K) {
        uint4 z = {0u, 0u, 0u, 0u};
        s4[0] = z; s4[1] = z;
      } else {
#pragma unroll
        for (int j = 0; j < 16; ++j) {
          int gk = kbeg + j;
          float v = (gk < K) ? B[(long long)gk * N + gn] : 0.f;
          Bs[bnc * 40 + bks + j] = (__bf16)v;
        }
      }
    }
    __syncthreads();

    FragA a0, a1, b0, b1;
    {
      const uint4* p = (const uint4*)&As[(wm + row16) * 40];
      a0.u[0] = p[kh];     a0.u[1] = p[2 + kh];
      const uint4* q = (const uint4*)&As[(wm + 16 + row16) * 40];
      a1.u[0] = q[kh];     a1.u[1] = q[2 + kh];
    }
    {
      const uint4* p = (const uint4*)&Bs[(wn + row16) * 40];
      b0.u[0] = p[kh];     b0.u[1] = p[2 + kh];
      const uint4* q = (const uint4*)&Bs[(wn + 16 + row16) * 40];
      b1.u[0] = q[kh];     b1.u[1] = q[2 + kh];
    }
    acc[0][0] = wmma_bf16(a0.v, b0.v, acc[0][0]);
    acc[0][1] = wmma_bf16(a0.v, b1.v, acc[0][1]);
    acc[1][0] = wmma_bf16(a1.v, b0.v, acc[1][0]);
    acc[1][1] = wmma_bf16(a1.v, b1.v, acc[1][1]);
    __syncthreads();
  }

#pragma unroll
  for (int ci = 0; ci < 2; ++ci) {
    const int cn = tn + wn + ci * 16 + row16;
    if (cn < N) {
      const float bv = bias ? bias[cn] : 0.f;
#pragma unroll
      for (int ri = 0; ri < 2; ++ri) {
#pragma unroll
        for (int r = 0; r < 8; ++r) {      // C layout: M = r + 8*kh
          int cm = tm + wm + ri * 16 + r + 8 * kh;
          if (cm < M) {
            float vvl = acc[ri][ci][r] + bv;
            if (R) vvl += R[(long long)cm * N + cn];
            C[(long long)cm * N + cn] = vvl;
          }
        }
      }
    }
  }
}

// ---------------------------------------------------------------------------
// Fused MHA: per (16-row i-tile, head, batch) one wave computes
// S = (Q@K^T)/8 -> masked softmax in LDS -> ctx = P@V; the [B,H,N,N]
// attention tensor (164 MB) never touches HBM. Q/K tiles are staged through
// LDS with float4 loads + packed b128 stores; fragments come from b128 LDS
// reads.
// ---------------------------------------------------------------------------
__global__ __launch_bounds__(32) void mha_attn_kernel(
    const float* __restrict__ q, const float* __restrict__ k,
    const float* __restrict__ v, const float* __restrict__ adj,
    float* __restrict__ ctx)
{
  __shared__ __align__(16) float srow[16][NPAD];   // 26624 B
  __shared__ __align__(16) __bf16 Ks[16 * 72];     // 2304 B (pitch 72 halves)
  const int i0 = blockIdx.x * 16, h = blockIdx.y, b = blockIdx.z;
  const int lane = threadIdx.x;
  const int row16 = lane & 15, kh = lane >> 4;
  const int rl = lane >> 1;            // staging row 0..15
  const int dseg = (lane & 1) * 32;    // staging d segment 0/32

  // ---- stage Q tile, build persistent A-fragments ----
  stage32(q + ((long long)(b * NNODE) + i0 + rl) * CDIM + h * HD + dseg,
          &Ks[rl * 72 + dseg]);
  __syncthreads();
  FragA qa0, qa1;
  {
    const uint4* p = (const uint4*)&Ks[row16 * 72];
    qa0.u[0] = p[kh];     qa0.u[1] = p[2 + kh];     // d = 0..31  (A layout)
    qa1.u[0] = p[4 + kh]; qa1.u[1] = p[6 + kh];     // d = 32..63
  }
  __syncthreads();

  // ---- scores: S = (Q @ K^T) / 8 ----
  for (int jt = 0; jt < 25; ++jt) {
    const int j0 = jt * 16;
    stage32(k + ((long long)(b * NNODE) + j0 + rl) * CDIM + h * HD + dseg,
            &Ks[rl * 72 + dseg]);
    __syncthreads();
    FragA kb0, kb1;
    {
      const uint4* p = (const uint4*)&Ks[row16 * 72];
      kb0.u[0] = p[kh];     kb0.u[1] = p[2 + kh];  // d = 0..31 (A layout)
      kb1.u[0] = p[4 + kh]; kb1.u[1] = p[6 + kh];  // d = 32..63 (A layout)
    }
    v8f acc = {0.f, 0.f, 0.f, 0.f, 0.f, 0.f, 0.f, 0.f};
    acc = wmma_bf16(qa0.v, kb0.v, acc);
    acc = wmma_bf16(qa1.v, kb1.v, acc);
#pragma unroll
    for (int r = 0; r < 8; ++r)
      srow[r + 8 * kh][j0 + row16] = acc[r] * 0.125f;   // 1/sqrt(64)
    __syncthreads();
  }
  if (lane < 16)
#pragma unroll
    for (int r = 0; r < 16; ++r) srow[r][NNODE + lane] = 0.f;
  __syncthreads();

  // ---- masked softmax over each of the 16 rows ----
  for (int r = 0; r < 16; ++r) {
    const float* arow = adj + ((long long)b * NNODE + (i0 + r)) * NNODE;
    float mx = -3.0e38f;
    for (int c = lane; c < NNODE; c += 32) {
      float e = srow[r][c];
      e = (arow[c] > 0.f) ? e : -1.0e9f;
      srow[r][c] = e;
      mx = fmaxf(mx, e);
    }
    mx = wred_max(mx);
    float sum = 0.f;
    for (int c = lane; c < NNODE; c += 32) {
      float p = __expf(srow[r][c] - mx);
      srow[r][c] = p;
      sum += p;
    }
    sum = wred_sum(sum);
    const float inv = 1.f / sum;
    for (int c = lane; c < NNODE; c += 32) srow[r][c] *= inv;
  }
  __syncthreads();

  // ---- ctx = P @ V : ks outer (pa reused across the 4 d-tiles) ----
  const int dbase = h * HD;
  const v8f vzz = {0.f, 0.f, 0.f, 0.f, 0.f, 0.f, 0.f, 0.f};
  v8f acc[4];
  acc[0] = vzz; acc[1] = vzz; acc[2] = vzz; acc[3] = vzz;
  for (int ksi = 0; ksi < 13; ++ksi) {
    const int jb = ksi * 32;
    v16bf pa{};
#pragma unroll
    for (int i = 0; i < 16; ++i) {
      int jl = jb + 16 * (i >> 3) + 8 * kh + (i & 7);
      pa[i] = (__bf16)srow[row16][jl];
    }
#pragma unroll
    for (int dt = 0; dt < 4; ++dt) {
      const int dcol = dbase + dt * 16 + row16;
      v16bf vb{};
#pragma unroll
      for (int i = 0; i < 16; ++i) {
        int j = jb + 16 * (i >> 3) + 8 * kh + (i & 7);
        float vvl = (j < NNODE) ? v[((long long)(b * NNODE) + j) * CDIM + dcol] : 0.f;
        vb[i] = (__bf16)vvl;
      }
      acc[dt] = wmma_bf16(pa, vb, acc[dt]);
    }
  }
#pragma unroll
  for (int dt = 0; dt < 4; ++dt) {
    const int dcol = dbase + dt * 16 + row16;
#pragma unroll
    for (int r = 0; r < 8; ++r) {
      int cm = i0 + r + 8 * kh;
      ctx[((long long)(b * NNODE) + cm) * CDIM + dcol] = acc[dt][r];
    }
  }
}

// GAT src/dst projections: src[row]=h[row]·a[:F], dst[row]=h[row]·a[F:]
__global__ __launch_bounds__(128) void srcdst_kernel(
    const float* __restrict__ h, const float* __restrict__ a,
    float* __restrict__ src, float* __restrict__ dst, int F, int rows)
{
  const int wave = threadIdx.x >> 5, lane = threadIdx.x & 31;
  const int row = blockIdx.x * 4 + wave;
  if (row >= rows) return;
  const float* hr = h + (long long)row * F;
  float s = 0.f, d = 0.f;
  for (int c = lane; c < F; c += 32) {
    float hv = hr[c];
    s += hv * a[c];
    d += hv * a[F + c];
  }
  s = wred_sum(s);
  d = wred_sum(d);
  if (lane == 0) { src[row] = s; dst[row] = d; }
}

// GAT edge scores + masked softmax, one wave per (b,i) row.
__global__ __launch_bounds__(32) void gat_row_kernel(
    const float* __restrict__ src, const float* __restrict__ dst,
    const float* __restrict__ adj, float* __restrict__ attn)
{
  const int row = blockIdx.x;                  // b*400 + i
  const int b = row / NNODE;
  const int lane = threadIdx.x;
  const float si = src[row];
  const float* dstb = dst + (long long)b * NNODE;
  const float* arow = adj + (long long)row * NNODE;
  float val[13];
  float mx = -3.0e38f;
#pragma unroll
  for (int t = 0; t < 13; ++t) {
    int j = lane + 32 * t;
    float e = -3.0e38f;
    if (j < NNODE) {
      e = si + dstb[j];
      e = (e > 0.f) ? e : 0.2f * e;            // leaky_relu(0.2)
      e = (arow[j] > 0.f) ? e : -1.0e9f;
    }
    val[t] = e;
    mx = fmaxf(mx, e);
  }
  mx = wred_max(mx);
  float sum = 0.f;
#pragma unroll
  for (int t = 0; t < 13; ++t) {
    int j = lane + 32 * t;
    float p = (j < NNODE) ? __expf(val[t] - mx) : 0.f;
    val[t] = p;
    sum += p;
  }
  sum = wred_sum(sum);
  const float inv = 1.f / sum;
  float* orow = attn + (long long)row * NNODE;
#pragma unroll
  for (int t = 0; t < 13; ++t) {
    int j = lane + 32 * t;
    if (j < NNODE) orow[j] = val[t] * inv;
  }
}

// LayerNorm (+ optional ELU), one wave per row.
__global__ __launch_bounds__(128) void ln_act_kernel(
    const float* __restrict__ in, const float* __restrict__ g,
    const float* __restrict__ bb, float* __restrict__ out,
    int F, int rows, int do_elu)
{
  const int wave = threadIdx.x >> 5, lane = threadIdx.x & 31;
  const int row = blockIdx.x * 4 + wave;
  if (row >= rows) return;
  const float* xr = in + (long long)row * F;
  float xv[8];
  const int cnt = F >> 5;
  float s = 0.f, s2 = 0.f;
  for (int t = 0; t < cnt; ++t) {
    float x = xr[lane + 32 * t];
    xv[t] = x;
    s += x;
    s2 += x * x;
  }
  s = wred_sum(s);
  s2 = wred_sum(s2);
  const float mean = s / F;
  const float var = s2 / F - mean * mean;
  const float rinv = rsqrtf(var + 1e-5f);
  for (int t = 0; t < cnt; ++t) {
    int c = lane + 32 * t;
    float y = (xv[t] - mean) * rinv * g[c] + bb[c];
    if (do_elu) y = (y > 0.f) ? y : (__expf(y) - 1.f);
    out[(long long)row * F + c] = y;
  }
}

extern "C" void kernel_launch(void* const* d_in, const int* in_sizes, int n_in,
                              void* d_out, int out_size, void* d_ws, size_t ws_size,
                              hipStream_t stream) {
  (void)in_sizes; (void)n_in; (void)out_size; (void)ws_size;
  const float* x    = (const float*)d_in[0];
  const float* adj  = (const float*)d_in[1];
  const float* Wq   = (const float*)d_in[2];
  const float* bq   = (const float*)d_in[3];
  const float* Wk   = (const float*)d_in[4];
  const float* bk   = (const float*)d_in[5];
  const float* Wv   = (const float*)d_in[6];
  const float* bv   = (const float*)d_in[7];
  const float* Wo   = (const float*)d_in[8];
  const float* bo   = (const float*)d_in[9];
  const float* g1W  = (const float*)d_in[10];
  const float* g1a  = (const float*)d_in[11];
  const float* g2W  = (const float*)d_in[12];
  const float* g2a  = (const float*)d_in[13];
  const float* g3W  = (const float*)d_in[14];
  const float* g3a  = (const float*)d_in[15];
  const float* r1W  = (const float*)d_in[16];
  const float* r1b  = (const float*)d_in[17];
  const float* r2W  = (const float*)d_in[18];
  const float* r2b  = (const float*)d_in[19];
  const float* r3W  = (const float*)d_in[20];
  const float* r3b  = (const float*)d_in[21];
  const float* ln1g = (const float*)d_in[22];
  const float* ln1b = (const float*)d_in[23];
  const float* ln2g = (const float*)d_in[24];
  const float* ln2b = (const float*)d_in[25];
  const float* ln3g = (const float*)d_in[26];
  const float* ln3b = (const float*)d_in[27];

  float* out = (float*)d_out;
  float* W   = (float*)d_ws;
  // scratch layout (floats)
  float* q   = W;                  // 25600*256
  float* kk  = W + 6553600;        // 25600*256
  float* vv  = W + 13107200;       // 25600*256
  float* ctx = W + 19660800;       // 25600*256
  float* xa  = W + 26214400;       // 25600*400
  float* hb  = W + 36454400;       // 25600*256 (reused per layer)
  float* rb  = W + 43008000;       // 25600*256 (reused per layer)
  float* xb  = W + 49561600;       // 25600*256 (x1/x2)
  float* src = W + 56115200;       // 25600
  float* dst = W + 56140800;       // 25600

  float* attn1 = out + 3276800;    // [64,400,400]
  float* attn2 = out + 13516800;
  float* attn3 = out + 23756800;

  const dim3 blk(128);

  // ---- MHA ----
  gemm_wmma<<<dim3(4, 400, 1), blk, 0, stream>>>(x, Wq, bq, nullptr, q,  25600, 256, 400, 0, 0, 0);
  gemm_wmma<<<dim3(4, 400, 1), blk, 0, stream>>>(x, Wk, bk, nullptr, kk, 25600, 256, 400, 0, 0, 0);
  gemm_wmma<<<dim3(4, 400, 1), blk, 0, stream>>>(x, Wv, bv, nullptr, vv, 25600, 256, 400, 0, 0, 0);
  mha_attn_kernel<<<dim3(25, 4, 64), dim3(32), 0, stream>>>(q, kk, vv, adj, ctx);
  // xa = x + ctx@Wo + bo
  gemm_wmma<<<dim3(7, 400, 1), blk, 0, stream>>>(ctx, Wo, bo, x, xa, 25600, 400, 256, 0, 0, 0);

  // ---- GAT layer 1 (400 -> 256) ----
  gemm_wmma<<<dim3(4, 400, 1), blk, 0, stream>>>(xa, g1W, nullptr, nullptr, hb, 25600, 256, 400, 0, 0, 0);
  srcdst_kernel<<<dim3(6400), blk, 0, stream>>>(hb, g1a, src, dst, 256, 25600);
  gat_row_kernel<<<dim3(25600), dim3(32), 0, stream>>>(src, dst, adj, attn1);
  gemm_wmma<<<dim3(4, 400, 1), blk, 0, stream>>>(xa, r1W, r1b, nullptr, rb, 25600, 256, 400, 0, 0, 0);
  gemm_wmma<<<dim3(4, 7, 64), blk, 0, stream>>>(attn1, hb, nullptr, rb, rb, 400, 256, 400, 160000, 102400, 102400);
  ln_act_kernel<<<dim3(6400), blk, 0, stream>>>(rb, ln1g, ln1b, xb, 256, 25600, 1);

  // ---- GAT layer 2 (256 -> 256) ----
  gemm_wmma<<<dim3(4, 400, 1), blk, 0, stream>>>(xb, g2W, nullptr, nullptr, hb, 25600, 256, 256, 0, 0, 0);
  srcdst_kernel<<<dim3(6400), blk, 0, stream>>>(hb, g2a, src, dst, 256, 25600);
  gat_row_kernel<<<dim3(25600), dim3(32), 0, stream>>>(src, dst, adj, attn2);
  gemm_wmma<<<dim3(4, 400, 1), blk, 0, stream>>>(xb, r2W, r2b, nullptr, rb, 25600, 256, 256, 0, 0, 0);
  gemm_wmma<<<dim3(4, 7, 64), blk, 0, stream>>>(attn2, hb, nullptr, rb, rb, 400, 256, 400, 160000, 102400, 102400);
  ln_act_kernel<<<dim3(6400), blk, 0, stream>>>(rb, ln2g, ln2b, xb, 256, 25600, 1);

  // ---- GAT layer 3 (256 -> 128), final LN without ELU ----
  gemm_wmma<<<dim3(2, 400, 1), blk, 0, stream>>>(xb, g3W, nullptr, nullptr, hb, 25600, 128, 256, 0, 0, 0);
  srcdst_kernel<<<dim3(6400), blk, 0, stream>>>(hb, g3a, src, dst, 128, 25600);
  gat_row_kernel<<<dim3(25600), dim3(32), 0, stream>>>(src, dst, adj, attn3);
  gemm_wmma<<<dim3(2, 400, 1), blk, 0, stream>>>(xb, r3W, r3b, nullptr, rb, 25600, 128, 256, 0, 0, 0);
  gemm_wmma<<<dim3(2, 7, 64), blk, 0, stream>>>(attn3, hb, nullptr, rb, rb, 400, 128, 400, 160000, 51200, 51200);
  ln_act_kernel<<<dim3(6400), blk, 0, stream>>>(rb, ln3g, ln3b, out, 128, 25600, 0);
}
